// DCMC_21534966022838
// MI455X (gfx1250) — matrix operand order repaired and checked by reading, not compiled, not run on hardware
//
#include <hip/hip_runtime.h>
#include <hip/hip_bf16.h>
#include <stdint.h>

// ---------------------------------------------------------------------------
// Types for CDNA5 WMMA (gfx1250, wave32): D(16x16 f32) = A(16x32 bf16) x B(32x16 bf16) + C
// ---------------------------------------------------------------------------
typedef __attribute__((ext_vector_type(16))) __bf16 bf16x16;
typedef __attribute__((ext_vector_type(8)))  __bf16 bf16x8;
typedef __attribute__((ext_vector_type(8)))  float  f32x8;

#define NB 4096          // batch
#define FD 256           // final feature dim
#define TEMP_INV 2.0f    // 1 / TEMP (TEMP = 0.5)

__device__ __forceinline__ __bf16 f2bf(float f) { return (__bf16)f; }

// A fragment (bf16 source): 16x32, lane L: row = row0 + (L&15), half = L>>4
//   element e<8  -> k = kk + 8*half + e          (contiguous, 16B)
//   element e>=8 -> k = kk + 16 + 8*half + (e-8) (contiguous, 16B)
__device__ __forceinline__ bf16x16 load_a_frag(const __bf16* __restrict__ A, int ld,
                                               int row0, int kk) {
    int lane = threadIdx.x & 31;
    int m = lane & 15, hl = lane >> 4;
    const __bf16* p = A + (size_t)(row0 + m) * ld + kk + 8 * hl;
    union { bf16x16 v; bf16x8 h[2]; } u;
    u.h[0] = *(const bf16x8*)(p);
    u.h[1] = *(const bf16x8*)(p + 16);
    return u.v;
}

// B fragment: 32x16 taken from BT (= B^T, [N x K] row-major, bf16).
// lane L: col n = n0 + (L&15), half = L>>4; element e<8 -> k = kk + 8*half + e, e>=8 -> k = kk + 16 + 8*half + (e-8)
__device__ __forceinline__ bf16x16 load_b_frag(const __bf16* __restrict__ BT, int ld,
                                               int n0, int kk) {
    int lane = threadIdx.x & 31;
    int n = lane & 15, hl = lane >> 4;
    const __bf16* p = BT + (size_t)(n0 + n) * ld + kk + 8 * hl;
    union { bf16x16 v; bf16x8 h[2]; } u;
    u.h[0] = *(const bf16x8*)(p);
    u.h[1] = *(const bf16x8*)(p + 16);
    return u.v;
}

__device__ __forceinline__ f32x8 wmma_bf16(bf16x16 a, bf16x16 b, f32x8 c) {
    return __builtin_amdgcn_wmma_f32_16x16x32_bf16(false, a, false, b,
                                                   (short)0, c, false, false);
}

__device__ __forceinline__ float wave_sum16(float v) {
    #pragma unroll
    for (int off = 1; off < 16; off <<= 1) v += __shfl_xor(v, off, 16);
    return v;
}

// ---------------------------------------------------------------------------
// GEMM: C[MxN] = A[MxK](bf16) @ BT[NxK](bf16)^T (+bias) (relu). Optional bf16 shadow Cb.
// block=128 (4 waves), each wave one 16x16 tile -> block covers 32x32.
// ---------------------------------------------------------------------------
__global__ void __launch_bounds__(128)
gemm_bt_kernel(float* __restrict__ C, const __bf16* __restrict__ A,
               const __bf16* __restrict__ BT, int M, int N, int K,
               const float* __restrict__ bias, int relu, __bf16* __restrict__ Cb) {
    int wave = threadIdx.x >> 5;
    int lane = threadIdx.x & 31;
    int hl = lane >> 4;
    int tm = blockIdx.y * 2 + (wave >> 1);
    int tn = blockIdx.x * 2 + (wave & 1);
    int row0 = tm * 16, col0 = tn * 16;

    f32x8 acc;
    #pragma unroll
    for (int i = 0; i < 8; ++i) acc[i] = 0.0f;

    for (int kk = 0; kk < K; kk += 32) {
        bf16x16 a = load_a_frag(A, K, row0, kk);
        bf16x16 b = load_b_frag(BT, K, col0, kk);
        acc = wmma_bf16(a, b, acc);
    }
    int col = col0 + (lane & 15);
    float bv = bias ? bias[col] : 0.0f;
    #pragma unroll
    for (int r = 0; r < 8; ++r) {
        int row = row0 + r + 8 * hl;
        float v = acc[r] + bv;
        if (relu) v = fmaxf(v, 0.0f);
        C[(size_t)row * N + col] = v;
        if (Cb) Cb[(size_t)row * N + col] = f2bf(v);
    }
}

// ---------------------------------------------------------------------------
// BatchNorm1d (training mode, biased var) over columns, in place; optional bf16 shadow.
// ---------------------------------------------------------------------------
__global__ void __launch_bounds__(256)
bn_kernel(float* __restrict__ X, int N, const float* __restrict__ gamma,
          const float* __restrict__ beta, int relu, __bf16* __restrict__ Xb) {
    int col = blockIdx.x, t = threadIdx.x;
    float s = 0.f, s2 = 0.f;
    for (int r = t; r < NB; r += 256) {
        float v = X[(size_t)r * N + col];
        s += v; s2 += v * v;
    }
    __shared__ float sh[256], sh2[256];
    sh[t] = s; sh2[t] = s2; __syncthreads();
    for (int off = 128; off > 0; off >>= 1) {
        if (t < off) { sh[t] += sh[t + off]; sh2[t] += sh2[t + off]; }
        __syncthreads();
    }
    float mu = sh[0] / (float)NB;
    float var = sh2[0] / (float)NB - mu * mu;
    float rstd = rsqrtf(var + 1e-5f);
    float g = gamma ? gamma[col] : 1.0f;
    float b = beta ? beta[col] : 0.0f;
    for (int r = t; r < NB; r += 256) {
        float v = X[(size_t)r * N + col];
        v = (v - mu) * rstd * g + b;
        if (relu) v = fmaxf(v, 0.0f);
        X[(size_t)r * N + col] = v;
        if (Xb) Xb[(size_t)r * N + col] = f2bf(v);
    }
}

// Row-wise L2 normalize [NB x 256]: f32 in -> bf16 out; one block per row.
__global__ void __launch_bounds__(256)
l2norm_kernel(const float* __restrict__ X, __bf16* __restrict__ Y) {
    int row = blockIdx.x, t = threadIdx.x;
    float v = X[(size_t)row * FD + t];
    __shared__ float sh[256];
    sh[t] = v * v; __syncthreads();
    for (int off = 128; off > 0; off >>= 1) {
        if (t < off) sh[t] += sh[t + off];
        __syncthreads();
    }
    float d = fmaxf(sqrtf(sh[0]), 1e-12f);
    Y[(size_t)row * FD + t] = f2bf(v / d);
}

// dst[n*K+k] (bf16) = ema? m*O[k*N+n]+(1-m)*T[k*N+n] : O[k*N+n]
__global__ void __launch_bounds__(256)
ema_transpose_kernel(__bf16* __restrict__ dst, const float* __restrict__ O,
                     const float* __restrict__ T, int K, int N,
                     const float* __restrict__ mom, int do_ema) {
    int idx = blockIdx.x * 256 + threadIdx.x;
    if (idx >= K * N) return;
    int k = idx / N, n = idx % N;
    float v = O[idx];
    if (do_ema) { float m = *mom; v = m * v + (1.0f - m) * T[idx]; }
    dst[(size_t)n * K + k] = f2bf(v);
}

__global__ void __launch_bounds__(256)
ema_vec_kernel(float* __restrict__ dst, const float* __restrict__ O,
               const float* __restrict__ T, int n, const float* __restrict__ mom) {
    int i = blockIdx.x * 256 + threadIdx.x;
    if (i < n) { float m = *mom; dst[i] = m * O[i] + (1.0f - m) * T[i]; }
}

__global__ void __launch_bounds__(256)
convert_bf16_kernel(const float* __restrict__ X, __bf16* __restrict__ Y, int n) {
    int i = blockIdx.x * 256 + threadIdx.x;
    if (i < n) Y[i] = f2bf(X[i]);
}

// ---------------------------------------------------------------------------
// Fused contrastive + adaptive-similarity-mask kernel (bf16 inputs).
//   pass 1: S_i = sum_j exp(-20*max(0,1-d_ij)); flash-softmax of logits; argmin_j |1-sim0_j|
//   pass 2: W_i = sum_{j!=i} exp(sim_ij); U_i = sum_j l_ij*sim_ij; capture l at argmin
// Closed form: sum_j sim = 1, diagonal weight == 1, only the argmin neighbor gets
// weight 1 - e^{sim_a}/W:   loss_row = (lse - U + corr) / (1 - thresh_a*sim_a*e^{sim_a}/W)
// ---------------------------------------------------------------------------
__global__ void __launch_bounds__(128)
contrastive_kernel(const __bf16* __restrict__ qn, const __bf16* __restrict__ kn,
                   float* __restrict__ partial, const int* __restrict__ warm_ptr) {
    int wave = threadIdx.x >> 5;
    int lane = threadIdx.x & 31;
    int hl = lane >> 4;
    int ln = lane & 15;
    int i0 = blockIdx.x * 16;
    int warm = warm_ptr[0];

    __shared__ float sbA[4][16], sbB[4][16];
    __shared__ int   sbI[4][16];
    __shared__ float fS[16], fLse[16], fBv[16], fLd[16], fW[16], fU[16], fLa[16];
    __shared__ int   fBi[16];
    __shared__ float rowloss[16];

    // Preload A fragments for this block's 16 rows (K=256 -> 8 k-steps)
    bf16x16 aq[8], ak[8];
    #pragma unroll
    for (int ks = 0; ks < 8; ++ks) {
        aq[ks] = load_a_frag(qn, FD, i0, ks * 32);
        ak[ks] = load_a_frag(kn, FD, i0, ks * 32);
    }

    float S[8], mx[8], ex[8], bv[8], ld[8]; int bi[8];
    #pragma unroll
    for (int r = 0; r < 8; ++r) {
        S[r] = 0.f; mx[r] = -3.0e38f; ex[r] = 0.f;
        bv[r] = 3.0e38f; bi[r] = 0x7fffffff; ld[r] = 0.f;
    }

    // ----- pass 1 -----
    for (int jt = wave; jt < NB / 16; jt += 4) {
        int jb = jt * 16;
        f32x8 dacc, lacc;
        #pragma unroll
        for (int r = 0; r < 8; ++r) { dacc[r] = 0.f; lacc[r] = 0.f; }
        #pragma unroll
        for (int ks = 0; ks < 8; ++ks) {
            bf16x16 b = load_b_frag(kn, FD, jb, ks * 32);
            dacc = wmma_bf16(ak[ks], b, dacc);
            lacc = wmma_bf16(aq[ks], b, lacc);
        }
        int jcol = jb + ln;
        #pragma unroll
        for (int r = 0; r < 8; ++r) {
            int irow = i0 + r + 8 * hl;
            float l = TEMP_INV * lacc[r];
            float nm = fmaxf(mx[r], l);
            ex[r] = ex[r] * __expf(mx[r] - nm) + __expf(l - nm);
            mx[r] = nm;
            if (jcol == irow) ld[r] = l;
            float d = (jcol == irow) ? 1.0f : dacc[r];
            float sim0 = __expf(-20.0f * fmaxf(1.0f - d, 0.0f));
            S[r] += sim0;
            float val = (jcol == irow) ? 3.0e38f : fabsf(1.0f - sim0);
            if (val < bv[r] || (val == bv[r] && jcol < bi[r])) { bv[r] = val; bi[r] = jcol; }
        }
    }

    // intra-wave reductions (16-lane groups)
    #pragma unroll
    for (int r = 0; r < 8; ++r) {
        S[r] = wave_sum16(S[r]);
        ld[r] = wave_sum16(ld[r]);
        #pragma unroll
        for (int off = 1; off < 16; off <<= 1) {
            float om = __shfl_xor(mx[r], off, 16);
            float oe = __shfl_xor(ex[r], off, 16);
            float nm = fmaxf(mx[r], om);
            ex[r] = ex[r] * __expf(mx[r] - nm) + oe * __expf(om - nm);
            mx[r] = nm;
        }
        #pragma unroll
        for (int off = 1; off < 16; off <<= 1) {
            float ov = __shfl_xor(bv[r], off, 16);
            int   oi = __shfl_xor(bi[r], off, 16);
            if (ov < bv[r] || (ov == bv[r] && oi < bi[r])) { bv[r] = ov; bi[r] = oi; }
        }
    }
    // cross-wave: S, ld
    if (ln == 0) {
        #pragma unroll
        for (int r = 0; r < 8; ++r) { sbA[wave][r + 8 * hl] = S[r]; sbB[wave][r + 8 * hl] = ld[r]; }
    }
    __syncthreads();
    if (threadIdx.x < 16) {
        int t = threadIdx.x;
        fS[t]  = sbA[0][t] + sbA[1][t] + sbA[2][t] + sbA[3][t];
        fLd[t] = sbB[0][t] + sbB[1][t] + sbB[2][t] + sbB[3][t];
    }
    __syncthreads();
    // cross-wave: (mx, ex) -> lse
    if (ln == 0) {
        #pragma unroll
        for (int r = 0; r < 8; ++r) { sbA[wave][r + 8 * hl] = mx[r]; sbB[wave][r + 8 * hl] = ex[r]; }
    }
    __syncthreads();
    if (threadIdx.x < 16) {
        int t = threadIdx.x;
        float m = sbA[0][t], e = sbB[0][t];
        for (int w = 1; w < 4; ++w) {
            float om = sbA[w][t], oe = sbB[w][t];
            float nm = fmaxf(m, om);
            e = e * __expf(m - nm) + oe * __expf(om - nm);
            m = nm;
        }
        fLse[t] = m + logf(e);
    }
    __syncthreads();
    // cross-wave: (bv, bi)
    if (ln == 0) {
        #pragma unroll
        for (int r = 0; r < 8; ++r) { sbA[wave][r + 8 * hl] = bv[r]; sbI[wave][r + 8 * hl] = bi[r]; }
    }
    __syncthreads();
    if (threadIdx.x < 16) {
        int t = threadIdx.x;
        float v = sbA[0][t]; int ii = sbI[0][t];
        for (int w = 1; w < 4; ++w) {
            float ov = sbA[w][t]; int oi = sbI[w][t];
            if (ov < v || (ov == v && oi < ii)) { v = ov; ii = oi; }
        }
        fBv[t] = v; fBi[t] = ii;
    }
    __syncthreads();

    // ----- pass 2 (skipped in warm-up: mask = identity) -----
    if (!warm) {
        float invS_r[8]; int bi_r[8];
        #pragma unroll
        for (int r = 0; r < 8; ++r) {
            invS_r[r] = 1.0f / fS[r + 8 * hl];
            bi_r[r]   = fBi[r + 8 * hl];
        }
        float W[8], U[8], la[8];
        #pragma unroll
        for (int r = 0; r < 8; ++r) { W[r] = 0.f; U[r] = 0.f; la[r] = 0.f; }

        for (int jt = wave; jt < NB / 16; jt += 4) {
            int jb = jt * 16;
            f32x8 dacc, lacc;
            #pragma unroll
            for (int r = 0; r < 8; ++r) { dacc[r] = 0.f; lacc[r] = 0.f; }
            #pragma unroll
            for (int ks = 0; ks < 8; ++ks) {
                bf16x16 b = load_b_frag(kn, FD, jb, ks * 32);
                dacc = wmma_bf16(ak[ks], b, dacc);
                lacc = wmma_bf16(aq[ks], b, lacc);
            }
            int jcol = jb + ln;
            #pragma unroll
            for (int r = 0; r < 8; ++r) {
                int irow = i0 + r + 8 * hl;
                float d = (jcol == irow) ? 1.0f : dacc[r];
                float sim = __expf(-20.0f * fmaxf(1.0f - d, 0.0f)) * invS_r[r];
                float l = TEMP_INV * lacc[r];
                if (jcol != irow) W[r] += __expf(sim);
                U[r] += l * sim;
                if (jcol == bi_r[r]) la[r] = l;
            }
        }
        #pragma unroll
        for (int r = 0; r < 8; ++r) {
            W[r] = wave_sum16(W[r]);
            U[r] = wave_sum16(U[r]);
            la[r] = wave_sum16(la[r]);
        }
        if (ln == 0) {
            #pragma unroll
            for (int r = 0; r < 8; ++r) { sbA[wave][r + 8 * hl] = W[r]; sbB[wave][r + 8 * hl] = U[r]; }
        }
        __syncthreads();
        if (threadIdx.x < 16) {
            int t = threadIdx.x;
            fW[t] = sbA[0][t] + sbA[1][t] + sbA[2][t] + sbA[3][t];
            fU[t] = sbB[0][t] + sbB[1][t] + sbB[2][t] + sbB[3][t];
        }
        __syncthreads();
        if (ln == 0) {
            #pragma unroll
            for (int r = 0; r < 8; ++r) sbA[wave][r + 8 * hl] = la[r];
        }
        __syncthreads();
        if (threadIdx.x < 16) {
            int t = threadIdx.x;
            fLa[t] = sbA[0][t] + sbA[1][t] + sbA[2][t] + sbA[3][t];
        }
        __syncthreads();
    }

    // ----- per-row scalar epilogue -----
    if (threadIdx.x < 16) {
        int t = threadIdx.x;
        float lse = fLse[t];
        float loss;
        if (warm) {
            loss = lse - fLd[t];                       // mask = eye
        } else {
            float invS = 1.0f / fS[t];
            float T = lse - fU[t];                     // sum_j (lse - l_j) * sim_j
            float sim_a = (1.0f - fBv[t]) * invS;      // sim at argmin neighbor
            float diff_a = fBv[t] * invS;              // |diag - sim_a|
            float rs = 1.0f, corr = 0.0f;
            if (diff_a < 0.7f) {
                float g = sim_a * __expf(sim_a) / fW[t];
                rs = 1.0f - g;
                corr = -(lse - fLa[t]) * g;
            }
            loss = (T + corr) / rs;
        }
        rowloss[t] = loss;
    }
    __syncthreads();
    if (threadIdx.x == 0) {
        float s = 0.f;
        for (int t = 0; t < 16; ++t) s += rowloss[t];
        partial[blockIdx.x] = s;
    }
}

// Fixed-order final reduction of 9 x 256 partials -> scalar loss
__global__ void __launch_bounds__(256)
final_reduce_kernel(const float* __restrict__ partial, float* __restrict__ out) {
    int t = threadIdx.x;
    __shared__ float sh[256];
    __shared__ float total;
    if (t == 0) total = 0.f;
    __syncthreads();
    const float denom = 1.0f / ((float)NB * (float)NB);
    for (int c = 0; c < 9; ++c) {
        sh[t] = partial[c * 256 + t];
        __syncthreads();
        for (int off = 128; off > 0; off >>= 1) {
            if (t < off) sh[t] += sh[t + off];
            __syncthreads();
        }
        if (t == 0) {
            float w = (c < 3) ? (1.0f / 3.0f) : (1.0f / 6.0f);
            total += sh[0] * denom * w;
        }
        __syncthreads();
    }
    if (t == 0) out[0] = total;
}

// ---------------------------------------------------------------------------
// Host launch orchestration
// ---------------------------------------------------------------------------
static void run_gemm(float* C, const __bf16* A, const __bf16* BT, int M, int N, int K,
                     const float* bias, int relu, __bf16* Cb, hipStream_t s) {
    dim3 grid(N / 32, M / 32);
    gemm_bt_kernel<<<grid, 128, 0, s>>>(C, A, BT, M, N, K, bias, relu, Cb);
}
static void run_bn(float* X, int N, const float* g, const float* b, int relu,
                   __bf16* Xb, hipStream_t s) {
    bn_kernel<<<N, 256, 0, s>>>(X, N, g, b, relu, Xb);
}
static void run_transpose(__bf16* dst, const float* O, const float* T, int K, int N,
                          const float* mom, int ema, hipStream_t s) {
    int total = K * N;
    ema_transpose_kernel<<<(total + 255) / 256, 256, 0, s>>>(dst, O, T, K, N, mom, ema);
}

extern "C" void kernel_launch(void* const* d_in, const int* in_sizes, int n_in,
                              void* d_out, int out_size, void* d_ws, size_t ws_size,
                              hipStream_t stream) {
    (void)in_sizes; (void)n_in; (void)out_size; (void)ws_size;

    // Input layout (jax tree-leaves order; dict keys sorted: Ws < betas < gammas; W1<W2<b1<b2)
    const float* data[3] = { (const float*)d_in[0], (const float*)d_in[1], (const float*)d_in[2] };
    #define ON(v, l) ((const float*)d_in[3  + (v)*7 + (l)])
    #define TG(v, l) ((const float*)d_in[24 + (v)*7 + (l)])
    #define DE(v, l) ((const float*)d_in[45 + (v)*4 + (l)])
    const float* mom  = (const float*)d_in[57];
    const int*   warm = (const int*)d_in[58];

    // Workspace carve-up (all sub-offsets keep >=32B alignment)
    const size_t WT_SZ = 1024*1024 + 512*1024 + 256*512;       // 1703936 elems per view
    __bf16* datab = (__bf16*)d_ws;                             // 3 * NB*1024 bf16
    __bf16* wt_on = datab + 3 * (size_t)NB * 1024;             // 3 * WT_SZ bf16 (EMA'd online, transposed)
    __bf16* wt_tg = wt_on + 3 * WT_SZ;                         // 3 * WT_SZ bf16 (target, transposed)
    __bf16* wdt   = wt_tg + 3 * WT_SZ;                         // 3 * 524288 bf16 (decoder, transposed)
    __bf16* fab   = wdt + 3 * 524288;                          // 3 * NB*FD bf16 (f_a, post-BN)
    __bf16* featb = fab + 3 * (size_t)NB * FD;                 // 9 * NB*FD bf16 (l2-normalized feats)
    __bf16* h1b   = featb + 9 * (size_t)NB * FD;               // NB*1024 bf16
    __bf16* h2b   = h1b + (size_t)NB * 1024;                   // NB*512 bf16
    float*  gb    = (float*)(h2b + (size_t)NB * 512);          // 3 * 3072 f32 (EMA'd gamma/beta)
    float*  feat  = gb + 3 * 3072;                             // 9 * NB*FD f32
    float*  h1    = feat + 9 * (size_t)NB * FD;                // NB*1024 f32
    float*  h2    = h1 + (size_t)NB * 1024;                    // NB*512 f32
    float*  part  = h2 + (size_t)NB * 512;                     // 9 * 256 f32

    // --- prep: bf16 data copies; EMA'd online weights / target weights / decoder (transposed bf16) ---
    for (int v = 0; v < 3; ++v) {
        convert_bf16_kernel<<<(NB * 1024 + 255) / 256, 256, 0, stream>>>(
            data[v], datab + (size_t)v * NB * 1024, NB * 1024);
        __bf16* wto = wt_on + v * WT_SZ;
        __bf16* wtt = wt_tg + v * WT_SZ;
        float*  gbv = gb + v * 3072;
        __bf16* wdv = wdt + (size_t)v * 524288;
        run_transpose(wto,           ON(v,0), TG(v,0), 1024, 1024, mom, 1, stream);
        run_transpose(wto + 1048576, ON(v,1), TG(v,1), 1024, 512,  mom, 1, stream);
        run_transpose(wto + 1572864, ON(v,2), TG(v,2), 512,  256,  mom, 1, stream);
        run_transpose(wtt,           TG(v,0), TG(v,0), 1024, 1024, mom, 0, stream);
        run_transpose(wtt + 1048576, TG(v,1), TG(v,1), 1024, 512,  mom, 0, stream);
        run_transpose(wtt + 1572864, TG(v,2), TG(v,2), 512,  256,  mom, 0, stream);
        ema_vec_kernel<<<4, 256, 0, stream>>>(gbv,        ON(v,5), TG(v,5), 1024, mom); // gamma0
        ema_vec_kernel<<<4, 256, 0, stream>>>(gbv + 1024, ON(v,3), TG(v,3), 1024, mom); // beta0
        ema_vec_kernel<<<2, 256, 0, stream>>>(gbv + 2048, ON(v,6), TG(v,6), 512,  mom); // gamma1
        ema_vec_kernel<<<2, 256, 0, stream>>>(gbv + 2560, ON(v,4), TG(v,4), 512,  mom); // beta1
        run_transpose(wdv,          DE(v,0), DE(v,0), 256, 1024, mom, 0, stream);       // W1d^T
        run_transpose(wdv + 262144, DE(v,1), DE(v,1), 1024, 256, mom, 0, stream);       // W2d^T
    }

    // --- forward: f_a (online EMA branch), Q = mlp(f_a), f_b (target branch) ---
    for (int v = 0; v < 3; ++v) {
        float* fa = feat + (size_t)(0 + v) * NB * FD;
        float* fb = feat + (size_t)(3 + v) * NB * FD;
        float* Q  = feat + (size_t)(6 + v) * NB * FD;
        const __bf16* db  = datab + (size_t)v * NB * 1024;
        const __bf16* wto = wt_on + v * WT_SZ;
        const __bf16* wtt = wt_tg + v * WT_SZ;
        const float*  gbv = gb + v * 3072;
        const __bf16* wdv = wdt + (size_t)v * 524288;
        __bf16* fabv = fab + (size_t)v * NB * FD;

        // FCN with EMA'd online params -> f_a (bf16 shadow for the MLP)
        run_gemm(h1, db, wto,             NB, 1024, 1024, nullptr, 0, nullptr, stream);
        run_bn(h1, 1024, gbv, gbv + 1024, 1, h1b, stream);
        run_gemm(h2, h1b, wto + 1048576,  NB, 512, 1024, nullptr, 0, nullptr, stream);
        run_bn(h2, 512, gbv + 2048, gbv + 2560, 1, h2b, stream);
        run_gemm(fa, h2b, wto + 1572864,  NB, 256, 512, nullptr, 0, nullptr, stream);
        run_bn(fa, 256, nullptr, nullptr, 0, fabv, stream);

        // Q = mlp(f_a); hidden written with fused bias+relu and bf16 shadow
        run_gemm(h1, fabv, wdv,           NB, 1024, 256, DE(v,2), 1, h1b, stream);
        run_gemm(Q, h1b, wdv + 262144,    NB, 256, 1024, DE(v,3), 0, nullptr, stream);

        // FCN with target params -> f_b
        run_gemm(h1, db, wtt,             NB, 1024, 1024, nullptr, 0, nullptr, stream);
        run_bn(h1, 1024, TG(v,5), TG(v,3), 1, h1b, stream);
        run_gemm(h2, h1b, wtt + 1048576,  NB, 512, 1024, nullptr, 0, nullptr, stream);
        run_bn(h2, 512, TG(v,6), TG(v,4), 1, h2b, stream);
        run_gemm(fb, h2b, wtt + 1572864,  NB, 256, 512, nullptr, 0, nullptr, stream);
        run_bn(fb, 256, nullptr, nullptr, 0, nullptr, stream);
    }

    // --- l2-normalize all 9 feature matrices into bf16 buffers ---
    for (int i = 0; i < 9; ++i)
        l2norm_kernel<<<NB, 256, 0, stream>>>(feat + (size_t)i * NB * FD,
                                              featb + (size_t)i * NB * FD);

    // --- 9 fused contrastive calls: slots 0-2 intra, 3-8 inter ---
    const __bf16* fan[3]; const __bf16* fbn[3]; const __bf16* Qn[3];
    for (int v = 0; v < 3; ++v) {
        fan[v] = featb + (size_t)(0 + v) * NB * FD;
        fbn[v] = featb + (size_t)(3 + v) * NB * FD;
        Qn[v]  = featb + (size_t)(6 + v) * NB * FD;
    }
    const __bf16* qs[9] = { fan[0], fan[1], fan[2], Qn[0], Qn[0], Qn[1], Qn[1], Qn[2], Qn[2] };
    const __bf16* kk[9] = { fbn[0], fbn[1], fbn[2], fbn[1], fbn[2], fbn[0], fbn[2], fbn[0], fbn[1] };
    for (int c = 0; c < 9; ++c)
        contrastive_kernel<<<NB / 16, 128, 0, stream>>>(qs[c], kk[c], part + (size_t)c * 256, warm);

    final_reduce_kernel<<<1, 256, 0, stream>>>(part, (float*)d_out);
}
